// Attention_661424963782
// MI455X (gfx1250) — matrix operand order repaired and checked by reading, not compiled, not run on hardware
//
#include <hip/hip_runtime.h>
#include <hip/hip_bf16.h>
#include <stdint.h>

typedef __bf16 bf16;
typedef __attribute__((ext_vector_type(16))) __bf16 v16bf;
typedef __attribute__((ext_vector_type(8)))  float  v8f;

#define DMODEL 4096
#define NB     2
#define SLEN   2048
#define NHEAD  32
#define NKV    8
#define HDIM   128
#define SEG    512

__device__ __forceinline__ bf16 f2bf(float f) {
  union { float f; uint32_t u; } v; v.f = f;
  uint32_t r = v.u + 0x7FFFu + ((v.u >> 16) & 1u);
  uint16_t h = (uint16_t)(r >> 16);
  return __builtin_bit_cast(bf16, h);
}
__device__ __forceinline__ float bf2f(bf16 b) {
  union { uint32_t u; float f; } v;
  v.u = ((uint32_t)__builtin_bit_cast(uint16_t, b)) << 16;
  return v.f;
}

// CDNA5 async global->LDS copy (ASYNCcnt-tracked, no VGPR staging).
// INST_OFFSET is added to BOTH the global address and the LDS address (ISA 8.2/4.4),
// so contiguous chunks need only one base address pair + immediate offsets.
__device__ __forceinline__ void async_ld_b128(const void* gptr, uint32_t lds) {
  asm volatile("global_load_async_to_lds_b128 %0, %1, off"
               :: "v"(lds), "v"(gptr) : "memory");
}
#define ALD(gp, lp, o)                                                      \
  asm volatile("global_load_async_to_lds_b128 %0, %1, off offset:" #o       \
               :: "v"(lp), "v"(gp) : "memory")
// 128 contiguous bytes (one 64-elem bf16 row chunk) via 8 async b128 with imm offsets
#define ASYNC_ROW128(gp, lp)                                                \
  do {                                                                      \
    ALD(gp, lp, 0);  ALD(gp, lp, 16); ALD(gp, lp, 32);  ALD(gp, lp, 48);    \
    ALD(gp, lp, 64); ALD(gp, lp, 80); ALD(gp, lp, 96);  ALD(gp, lp, 112);   \
  } while (0)

__device__ __forceinline__ void wait_async0() {
  asm volatile("s_wait_asynccnt 0x0" ::: "memory");
}
__device__ __forceinline__ uint32_t lds_off(const void* p) {
  return (uint32_t)(uintptr_t)p;   // low 32 bits of generic LDS addr = LDS byte offset
}

// ---------------------------------------------------------------- convert (8 elems/thread)
__global__ void cvt_f32_bf16_v8(const float* __restrict__ in, bf16* __restrict__ out, int n8) {
  int i = blockIdx.x * blockDim.x + threadIdx.x;
  if (i >= n8) return;
  const float4* in4 = (const float4*)in;
  float4 a = in4[2 * i], b = in4[2 * i + 1];
  union { bf16 h[8]; uint4 u; } o;
  o.h[0] = f2bf(a.x); o.h[1] = f2bf(a.y); o.h[2] = f2bf(a.z); o.h[3] = f2bf(a.w);
  o.h[4] = f2bf(b.x); o.h[5] = f2bf(b.y); o.h[6] = f2bf(b.z); o.h[7] = f2bf(b.w);
  ((uint4*)out)[i] = o.u;
}

// fp32 [K][N] row-major -> bf16 [N][K] (transposed), LDS-tiled 32x32
__global__ __launch_bounds__(256) void cvt_transpose_f32_bf16(const float* __restrict__ in,
                                                              bf16* __restrict__ out,
                                                              int K, int N) {
  __shared__ float tile[32][33];
  const int nt = blockIdx.x * 32;
  const int kt = blockIdx.y * 32;
  const int tx = threadIdx.x & 31;
  const int ty = threadIdx.x >> 5;   // 0..7
#pragma unroll
  for (int i = 0; i < 32; i += 8)
    tile[ty + i][tx] = in[(size_t)(kt + ty + i) * N + nt + tx];
  __syncthreads();
#pragma unroll
  for (int i = 0; i < 32; i += 8)
    out[(size_t)(nt + ty + i) * K + kt + tx] = f2bf(tile[tx][ty + i]);
}

// ---------------------------------------------------------------- GEMM
// C[M,N] = A[M,K] @ B[K,N] with B supplied TRANSPOSED: Bt[N][K].
// Block 128x128, K-step 32, 8 waves, async double-buffered LDS tiles.
template <bool OUT_BF16>
__global__ __launch_bounds__(256) void gemm_wmma(const bf16* __restrict__ A,
                                                 const bf16* __restrict__ Bt,
                                                 void* __restrict__ Cp,
                                                 int M, int N, int K) {
  __shared__ bf16 As[2][128][40];   // [buf][m][k]
  __shared__ bf16 Bs[2][128][40];   // [buf][n][k]

  const int tid  = threadIdx.x;
  const int lane = tid & 31;
  const int wid  = tid >> 5;
  const int wm   = wid & 3;
  const int wn   = wid >> 2;
  const int l16  = lane & 15;
  const int lhi  = lane >> 4;

  const int m0 = blockIdx.y * 128;
  const int n0 = blockIdx.x * 128;

  const int lr = tid >> 2;         // 0..63
  const int lc = (tid & 3) * 8;    // 0,8,16,24

  // four global streams, advanced 64B per k-step
  const bf16* gA0 = A  + (size_t)(m0 + lr) * K + lc;
  const bf16* gA1 = gA0 + (size_t)64 * K;
  const bf16* gB0 = Bt + (size_t)(n0 + lr) * K + lc;
  const bf16* gB1 = gB0 + (size_t)64 * K;

  // LDS destinations for both double-buffer slots (computed once)
  uint32_t lA0[2], lA1[2], lB0[2], lB1[2];
#pragma unroll
  for (int bfi = 0; bfi < 2; ++bfi) {
    lA0[bfi] = lds_off(&As[bfi][lr][lc]);
    lA1[bfi] = lds_off(&As[bfi][lr + 64][lc]);
    lB0[bfi] = lds_off(&Bs[bfi][lr][lc]);
    lB1[bfi] = lds_off(&Bs[bfi][lr + 64][lc]);
  }

  v8f acc[2][4];
#pragma unroll
  for (int mi = 0; mi < 2; ++mi)
#pragma unroll
    for (int ns = 0; ns < 4; ++ns)
      acc[mi][ns] = (v8f){0.f, 0.f, 0.f, 0.f, 0.f, 0.f, 0.f, 0.f};

  // prologue: fill buffer 0
  async_ld_b128(gA0, lA0[0]);
  async_ld_b128(gA1, lA1[0]);
  async_ld_b128(gB0, lB0[0]);
  async_ld_b128(gB1, lB1[0]);
  wait_async0();
  __syncthreads();

  int cur = 0;
  for (int k0 = 0; k0 < K; k0 += 32) {
    const int nxt = cur ^ 1;
    if (k0 + 32 < K) {
      gA0 += 32; gA1 += 32; gB0 += 32; gB1 += 32;
      async_ld_b128(gA0, lA0[nxt]);
      async_ld_b128(gA1, lA1[nxt]);
      async_ld_b128(gB0, lB0[nxt]);
      async_ld_b128(gB1, lB1[nxt]);
    }

    // A fragments (16-bit A 16x32: lanes0-15 K0-7/16-23, lanes16-31 K8-15/24-31)
    v16bf afr[2];
#pragma unroll
    for (int mi = 0; mi < 2; ++mi) {
      int m = wm * 32 + mi * 16 + l16;
      int koff = lhi * 8;
      union { v16bf v; uint32_t u[8]; } t;
#pragma unroll
      for (int q = 0; q < 8; ++q) {
        int kk = 2 * q + ((q >= 4) ? 8 : 0) + koff;
        t.u[q] = *(const uint32_t*)&As[cur][m][kk];
      }
      afr[mi] = t.v;
    }
#pragma unroll
    for (int ns = 0; ns < 4; ++ns) {
      // B fragment from transposed tile: per-lane K in table order (matches A)
      int n = wn * 64 + ns * 16 + l16;
      union { v16bf v; uint32_t u[8]; } tb;
#pragma unroll
      for (int q = 0; q < 8; ++q)
        tb.u[q] = *(const uint32_t*)&Bs[cur][n][lhi * 8 + 2 * q + ((q >= 4) ? 8 : 0)];
#pragma unroll
      for (int mi = 0; mi < 2; ++mi)
        acc[mi][ns] = __builtin_amdgcn_wmma_f32_16x16x32_bf16(
            false, afr[mi], false, tb.v, (short)0, acc[mi][ns], false, false);
    }

    wait_async0();
    __syncthreads();
    cur = nxt;
  }

#pragma unroll
  for (int mi = 0; mi < 2; ++mi)
#pragma unroll
    for (int ns = 0; ns < 4; ++ns)
#pragma unroll
      for (int r = 0; r < 8; ++r) {
        int row = m0 + wm * 32 + mi * 16 + r + 8 * lhi;
        int col = n0 + wn * 64 + ns * 16 + l16;
        float val = acc[mi][ns][r];
        if (OUT_BF16) ((bf16*)Cp)[(size_t)row * N + col] = f2bf(val);
        else          ((float*)Cp)[(size_t)row * N + col] = val;
      }
}

// ---------------------------------------------------------------- RoPE (in-place, bf16)
__global__ void rope_kernel(bf16* __restrict__ t, const float* __restrict__ cosb,
                            const float* __restrict__ sinb, int heads) {
  int idx = blockIdx.x * blockDim.x + threadIdx.x;
  int total = NB * SLEN * heads * (HDIM / 2);
  if (idx >= total) return;
  int p   = idx & 63;
  int h   = (idx >> 6) % heads;
  int row = idx / (64 * heads);
  int s   = row & (SLEN - 1);
  float c  = cosb[s * 64 + p];
  float sn = sinb[s * 64 + p];
  size_t base = (size_t)row * heads * HDIM + h * HDIM + 2 * p;
  float e = bf2f(t[base]);
  float o = bf2f(t[base + 1]);
  t[base]     = f2bf(e * c - o * sn);
  t[base + 1] = f2bf(e * sn + o * c);
}

// ---------------------------------------------------------------- flash attention
// one block = (b, head, 64 q rows); 4 waves, 16 q rows each.
__global__ __launch_bounds__(128) void flash_attn(const bf16* __restrict__ Qb,
                                                  const bf16* __restrict__ Kb,
                                                  const bf16* __restrict__ Vb,
                                                  bf16* __restrict__ Ob) {
  __shared__ bf16 Qs[64][136];
  __shared__ bf16 Ks[64][136];
  __shared__ bf16 Vs[64][136];
  __shared__ bf16 Ps[64][72];

  const int tid  = threadIdx.x;
  const int lane = tid & 31;
  const int wvid = tid >> 5;
  const int l16  = lane & 15;
  const int lhi  = lane >> 4;

  const int q0  = blockIdx.x * 64;
  const int h   = blockIdx.y;
  const int b   = blockIdx.z;
  const int kvh = h >> 2;   // REP = 4

  const int lr  = tid >> 1;        // 0..63
  const int lc0 = (tid & 1) * 64;  // 0 or 64

  const int   seg0 = (q0 / SEG) * SEG;
  const size_t kvstride = (size_t)64 * NKV * HDIM;   // one key-block of rows

  { // async load Q tile 64x128 (one base + imm offsets)
    const bf16* g = Qb + (size_t)(b * SLEN + q0 + lr) * (NHEAD * HDIM) + h * HDIM + lc0;
    uint32_t l = lds_off(&Qs[lr][lc0]);
    ASYNC_ROW128(g, l);
  }
  wait_async0();
  __syncthreads();

  // hoist Q A-fragments: 4 chunks of K=32 over HD=128
  v16bf aq[4];
  {
    int m = wvid * 16 + l16;
    int koff = lhi * 8;
#pragma unroll
    for (int kc = 0; kc < 4; ++kc) {
      union { v16bf v; uint32_t u[8]; } t;
#pragma unroll
      for (int q = 0; q < 8; ++q) {
        int kk = kc * 32 + 2 * q + ((q >= 4) ? 8 : 0) + koff;
        t.u[q] = *(const uint32_t*)&Qs[m][kk];
      }
      aq[kc] = t.v;
    }
  }

  float mrow[8], lrow[8];
  v8f oacc[8];
#pragma unroll
  for (int r = 0; r < 8; ++r) { mrow[r] = -1e30f; lrow[r] = 0.f; }
#pragma unroll
  for (int d = 0; d < 8; ++d)
    oacc[d] = (v8f){0.f, 0.f, 0.f, 0.f, 0.f, 0.f, 0.f, 0.f};

  const float scale = 0.08838834764831845f; // 1/sqrt(128)
  const float L2E   = 1.44269504088896f;

  // K/V stream pointers (advanced one key-block per iteration)
  const bf16* gk = Kb + (size_t)(b * SLEN + seg0 + lr) * (NKV * HDIM) + kvh * HDIM + lc0;
  const bf16* gv = Vb + (size_t)(b * SLEN + seg0 + lr) * (NKV * HDIM) + kvh * HDIM + lc0;
  const uint32_t lK = lds_off(&Ks[lr][lc0]);
  const uint32_t lV = lds_off(&Vs[lr][lc0]);

  for (int kb = seg0; kb <= q0; kb += 64) {
    ASYNC_ROW128(gk, lK);
    ASYNC_ROW128(gv, lV);
    gk += kvstride;
    gv += kvstride;
    wait_async0();
    __syncthreads();

    // ---- scores: S(16x64) = Q(16x128) . K^T ----
    v8f sacc[4];
#pragma unroll
    for (int ns = 0; ns < 4; ++ns)
      sacc[ns] = (v8f){0.f, 0.f, 0.f, 0.f, 0.f, 0.f, 0.f, 0.f};
#pragma unroll
    for (int ns = 0; ns < 4; ++ns) {
#pragma unroll
      for (int kc = 0; kc < 4; ++kc) {
        union { v16bf v; uint32_t u[8]; } tb;
#pragma unroll
        for (int q = 0; q < 8; ++q)
          tb.u[q] = *(const uint32_t*)&Ks[ns * 16 + l16][kc * 32 + lhi * 8 + 2 * q + ((q >= 4) ? 8 : 0)];
        sacc[ns] = __builtin_amdgcn_wmma_f32_16x16x32_bf16(
            false, aq[kc], false, tb.v, (short)0, sacc[ns], false, false);
      }
    }

    // ---- scale + causal mask (diagonal block only) ----
    const bool diag = (kb == q0);
#pragma unroll
    for (int ns = 0; ns < 4; ++ns) {
      int key = kb + ns * 16 + l16;
#pragma unroll
      for (int r = 0; r < 8; ++r) {
        float v = sacc[ns][r] * scale;
        if (diag) {
          int qrow = q0 + wvid * 16 + r + 8 * lhi;
          if (key > qrow) v = -1e30f;
        }
        sacc[ns][r] = v;
      }
    }

    // ---- online softmax per row (rows striped over 16-lane groups) ----
#pragma unroll
    for (int r = 0; r < 8; ++r) {
      float mx = fmaxf(fmaxf(sacc[0][r], sacc[1][r]), fmaxf(sacc[2][r], sacc[3][r]));
#pragma unroll
      for (int d = 1; d < 16; d <<= 1) mx = fmaxf(mx, __shfl_xor(mx, d, 32));
      float mnew = fmaxf(mrow[r], mx);
      float corr = exp2f((mrow[r] - mnew) * L2E);
      float sum = 0.f;
#pragma unroll
      for (int ns = 0; ns < 4; ++ns) {
        float p = exp2f((sacc[ns][r] - mnew) * L2E);
        sacc[ns][r] = p;
        sum += p;
      }
#pragma unroll
      for (int d = 1; d < 16; d <<= 1) sum += __shfl_xor(sum, d, 32);
      lrow[r] = lrow[r] * corr + sum;
      mrow[r] = mnew;
#pragma unroll
      for (int dd = 0; dd < 8; ++dd) oacc[dd][r] *= corr;
    }

    // ---- restripe P into A-matrix layout via LDS (same-wave region) ----
#pragma unroll
    for (int ns = 0; ns < 4; ++ns)
#pragma unroll
      for (int r = 0; r < 8; ++r)
        Ps[wvid * 16 + r + 8 * lhi][ns * 16 + l16] = f2bf(sacc[ns][r]);
    asm volatile("s_wait_dscnt 0x0" ::: "memory");  // same-wave LDS RAW ordering

    v16bf ap[2];
#pragma unroll
    for (int kc = 0; kc < 2; ++kc) {
      int m = wvid * 16 + l16;
      int koff = lhi * 8;
      union { v16bf v; uint32_t u[8]; } t;
#pragma unroll
      for (int q = 0; q < 8; ++q) {
        int kk = kc * 32 + 2 * q + ((q >= 4) ? 8 : 0) + koff;
        t.u[q] = *(const uint32_t*)&Ps[m][kk];
      }
      ap[kc] = t.v;
    }

    // ---- O += P(16x64) . V(64x128) ----
#pragma unroll
    for (int ds = 0; ds < 8; ++ds) {
#pragma unroll
      for (int kc = 0; kc < 2; ++kc) {
        union { v16bf v; uint16_t h[16]; } tb;
#pragma unroll
        for (int hh = 0; hh < 16; ++hh)
          tb.h[hh] = *(const uint16_t*)&Vs[kc * 32 + lhi * 8 + hh + ((hh >= 8) ? 8 : 0)][ds * 16 + l16];
        oacc[ds] = __builtin_amdgcn_wmma_f32_16x16x32_bf16(
            false, ap[kc], false, tb.v, (short)0, oacc[ds], false, false);
      }
    }
    __syncthreads();
  }

  // ---- normalize and store ----
#pragma unroll
  for (int r = 0; r < 8; ++r) {
    float inv = 1.0f / lrow[r];
    int row = b * SLEN + q0 + wvid * 16 + r + 8 * lhi;
#pragma unroll
    for (int ds = 0; ds < 8; ++ds) {
      int col = h * HDIM + ds * 16 + l16;
      Ob[(size_t)row * (NHEAD * HDIM) + col] = f2bf(oacc[ds][r] * inv);
    }
  }
}

// ---------------------------------------------------------------- launch
extern "C" void kernel_launch(void* const* d_in, const int* in_sizes, int n_in,
                              void* d_out, int out_size, void* d_ws, size_t ws_size,
                              hipStream_t stream) {
  const float* x  = (const float*)d_in[0];
  const float* fc = (const float*)d_in[1];
  const float* fs = (const float*)d_in[2];
  const float* wq = (const float*)d_in[3];
  const float* wk = (const float*)d_in[4];
  const float* wv = (const float*)d_in[5];
  const float* wo = (const float*)d_in[6];
  float* out = (float*)d_out;

  const size_t MR  = (size_t)NB * SLEN;   // 4096 rows
  const size_t DKV = (size_t)NKV * HDIM;  // 1024

  char* ws = (char*)d_ws;
  size_t off = 0;
  auto carve = [&](size_t bytes) -> void* {
    void* p = ws + off;
    off += (bytes + 255) & ~(size_t)255;
    return p;
  };
  bf16* xb   = (bf16*)carve(MR * DMODEL * 2);
  bf16* wqbT = (bf16*)carve((size_t)DMODEL * DMODEL * 2);  // [N=4096][K=4096]
  bf16* wkbT = (bf16*)carve((size_t)DKV * DMODEL * 2);     // [N=1024][K=4096]
  bf16* wvbT = (bf16*)carve((size_t)DKV * DMODEL * 2);
  bf16* wobT = (bf16*)carve((size_t)DMODEL * DMODEL * 2);
  bf16* qb   = (bf16*)carve(MR * DMODEL * 2);
  bf16* kb   = (bf16*)carve(MR * DKV * 2);
  bf16* vb   = (bf16*)carve(MR * DKV * 2);
  bf16* ab   = (bf16*)carve(MR * DMODEL * 2);

  // x: plain convert (stays row-major A matrix)
  {
    size_t n8 = (MR * DMODEL) / 8;
    cvt_f32_bf16_v8<<<dim3((unsigned)((n8 + 255) / 256)), 256, 0, stream>>>(x, xb, (int)n8);
  }
  // weights: convert + transpose -> [N][K]
  cvt_transpose_f32_bf16<<<dim3(DMODEL / 32, DMODEL / 32), 256, 0, stream>>>(wq, wqbT, DMODEL, DMODEL);
  cvt_transpose_f32_bf16<<<dim3(DKV / 32,    DMODEL / 32), 256, 0, stream>>>(wk, wkbT, DMODEL, (int)DKV);
  cvt_transpose_f32_bf16<<<dim3(DKV / 32,    DMODEL / 32), 256, 0, stream>>>(wv, wvbT, DMODEL, (int)DKV);
  cvt_transpose_f32_bf16<<<dim3(DMODEL / 32, DMODEL / 32), 256, 0, stream>>>(wo, wobT, DMODEL, DMODEL);

  // projections (C = A @ B, B given transposed)
  gemm_wmma<true><<<dim3(DMODEL / 128, MR / 128), 256, 0, stream>>>(xb, wqbT, qb, (int)MR, DMODEL, DMODEL);
  gemm_wmma<true><<<dim3(DKV / 128,    MR / 128), 256, 0, stream>>>(xb, wkbT, kb, (int)MR, (int)DKV, DMODEL);
  gemm_wmma<true><<<dim3(DKV / 128,    MR / 128), 256, 0, stream>>>(xb, wvbT, vb, (int)MR, (int)DKV, DMODEL);

  // RoPE
  {
    size_t nq = MR * NHEAD * (HDIM / 2);
    size_t nk = MR * NKV * (HDIM / 2);
    rope_kernel<<<dim3((unsigned)((nq + 255) / 256)), 256, 0, stream>>>(qb, fc, fs, NHEAD);
    rope_kernel<<<dim3((unsigned)((nk + 255) / 256)), 256, 0, stream>>>(kb, fc, fs, NKV);
  }

  // flash attention
  flash_attn<<<dim3(SLEN / 64, NHEAD, NB), 128, 0, stream>>>(qb, kb, vb, ab);

  // output projection (f32 out)
  gemm_wmma<false><<<dim3(DMODEL / 128, MR / 128), 256, 0, stream>>>(ab, wobT, out, (int)MR, DMODEL, DMODEL);
}
